// LocMotionAppearanceSigned_33457795235935
// MI455X (gfx1250) — matrix operand order repaired and checked by reading, not compiled, not run on hardware
//
#include <hip/hip_runtime.h>
#include <hip/hip_bf16.h>

// ---------------------------------------------------------------------------
// LocMotionAppearanceSigned GNN for MI455X (gfx1250, wave32).
// Pooling is the memory-roofline term (~245MB -> ~11us @ 23.3TB/s); GEMMs are
// tiny (~2 GFLOP) and run on v_wmma_f32_16x16x32_bf16 with LDS-staged tiles:
// batched coalesced global_load_b32 -> bf16 in LDS -> ds_load_b128 fragments.
// ---------------------------------------------------------------------------

typedef __attribute__((ext_vector_type(16))) __bf16 v16bf;
typedef __attribute__((ext_vector_type(8)))  float  v8f;

static constexpr int BATCH = 4, WID = 256, HEI = 256, NNODE = 4096;

#define DEVINL __device__ __forceinline__

// ---------------- utility kernels ----------------
__global__ void k_fill(float* p, float v, long n) {
  long i = (long)blockIdx.x * blockDim.x + threadIdx.x;
  if (i < n) p[i] = v;
}

__global__ void k_copy_i32(const int* __restrict__ s, int* __restrict__ d, long n) {
  long i = (long)blockIdx.x * blockDim.x + threadIdx.x;
  if (i < n) d[i] = s[i];
}

// ---------------- superpixel pooling (segment sums via L2 atomics) ----------
__global__ void k_pool(const int* __restrict__ labels, const float* __restrict__ fx,
                       const float* __restrict__ fy, const float* __restrict__ s0,
                       const float* __restrict__ s1, const float* __restrict__ s2,
                       float* coords, float* p0, float* p1, float* p2, float* cnt) {
  const int plane = WID * HEI;
  int idx = blockIdx.x * blockDim.x + threadIdx.x;
  if (idx >= BATCH * plane) return;
  int b = idx / plane, rem = idx - b * plane;
  int w = rem / HEI, h = rem - w * HEI;
  int l = labels[idx];
  atomicAdd(cnt + l, 1.0f);
  atomicAdd(coords + l * 4 + 0, (float)w * (1.0f / (WID - 1)));
  atomicAdd(coords + l * 4 + 1, (float)h * (1.0f / (HEI - 1)));
  atomicAdd(coords + l * 4 + 2, fx[idx]);
  atomicAdd(coords + l * 4 + 3, fy[idx]);
  const float* s0b = s0 + (long)b * 32 * plane + rem;
  for (int c = 0; c < 32; ++c)  atomicAdd(p0 + l * 32 + c,  s0b[(long)c * plane]);
  const float* s1b = s1 + (long)b * 64 * plane + rem;
  for (int c = 0; c < 64; ++c)  atomicAdd(p1 + l * 64 + c,  s1b[(long)c * plane]);
  const float* s2b = s2 + (long)b * 128 * plane + rem;
  for (int c = 0; c < 128; ++c) atomicAdd(p2 + l * 128 + c, s2b[(long)c * plane]);
}

__global__ void k_pool_div(float* coords, float* p0, float* p1, float* p2,
                           const float* __restrict__ cnt) {
  int n = blockIdx.x * blockDim.x + threadIdx.x;
  if (n >= NNODE) return;
  float inv = 1.0f / fmaxf(cnt[n], 1.0f);
  for (int c = 0; c < 4;   ++c) coords[n * 4 + c]   *= inv;
  for (int c = 0; c < 32;  ++c) p0[n * 32 + c]      *= inv;
  for (int c = 0; c < 64;  ++c) p1[n * 64 + c]      *= inv;
  for (int c = 0; c < 128; ++c) p2[n * 128 + c]     *= inv;
}

// ---------------- WMMA bf16 GEMM: Out[M,N] (+)= A[M,K] @ Wt[K,N] (+bias) ----
// Block = 256 threads (8 waves) computes a 64x32 output tile (4x2 wmma tiles).
// Staging is fully static: each thread owns fixed (k | col) and 8|4 rows in an
// arithmetic progression, loads batched into registers (one wait), then
// mask-convert-store to LDS. Fragments read back as ds_load_b128. Epilogue
// accumulate loads are batched behind a single wait as well.
static constexpr int TM = 64, TN = 32, TK = 32;

__global__ void __launch_bounds__(256)
k_gemm_wmma(const float* __restrict__ A, int lda,
            const float* __restrict__ Wt, int ldw,
            const float* __restrict__ bias,
            float* __restrict__ Out, int ldo,
            int M, int K, int N, int acc, int relu) {
  __shared__ __align__(16) __bf16 As[TM * TK];   // [row][k]
  __shared__ __align__(16) __bf16 Bst[TN * TK];  // [col][k] (transposed)

  int tilesN = N / TN;
  int bM = blockIdx.x / tilesN, bN = blockIdx.x - bM * tilesN;
  int rowBase = bM * TM, colBase = bN * TN;
  int t = threadIdx.x;
  int waveId = t >> 5, lane = t & 31;
  int waveM = waveId >> 1, waveN = waveId & 1;   // 4 x 2 wmma tiles
  int half = lane >> 4, l16 = lane & 15;

  // static staging ownership (32-bit offsets only)
  const int ka  = t & 31;        // A: fixed k within tile
  const int ra0 = t >> 5;        // A: rows ra0 + 8j, j=0..7
  const int cb  = t & 31;        // B: fixed col within tile
  const int kb0 = t >> 5;        // B: k rows kb0 + 8j, j=0..3

  v8f cfrag = {};
  for (int k0 = 0; k0 < K; k0 += TK) {
    // ---- phase 1: batched global loads into registers ----
    int kkA = k0 + ka;
    int kcA = min(kkA, K - 1);
    float mA = (kkA < K) ? 1.0f : 0.0f;
    int offA = (rowBase + ra0) * lda + kcA;      // advances by 8*lda per j
    float va[8];
#pragma unroll
    for (int j = 0; j < 8; ++j) va[j] = A[offA + j * 8 * lda];

    float vb[4], mB[4];
#pragma unroll
    for (int j = 0; j < 4; ++j) {
      int kk = k0 + kb0 + 8 * j;
      mB[j] = (kk < K) ? 1.0f : 0.0f;
      vb[j] = Wt[min(kk, K - 1) * ldw + (colBase + cb)];
    }

    // ---- phase 2: mask, convert, store to LDS ----
#pragma unroll
    for (int j = 0; j < 8; ++j)
      As[(ra0 + 8 * j) * TK + ka] = (__bf16)(va[j] * mA);
#pragma unroll
    for (int j = 0; j < 4; ++j)
      Bst[cb * TK + kb0 + 8 * j] = (__bf16)(vb[j] * mB[j]);
    __syncthreads();

    // ---- fragments from LDS (ds_load_b128 x4 per lane) ----
    union { v16bf v; uint4 q[2]; } fa, fb;
    const __bf16* arow = As + (waveM * 16 + l16) * TK + half * 8;
    fa.q[0] = *(const uint4*)(arow);
    fa.q[1] = *(const uint4*)(arow + 16);
    const __bf16* bcol = Bst + (waveN * 16 + l16) * TK + half * 8;
    fb.q[0] = *(const uint4*)(bcol);
    fb.q[1] = *(const uint4*)(bcol + 16);
    cfrag = __builtin_amdgcn_wmma_f32_16x16x32_bf16(false, fa.v, false, fb.v,
                                                    (short)0, cfrag, false, false);
    __syncthreads();
  }

  // ---- epilogue: C/D layout (VGPR j -> M=j, +8 for hi half) ----
  int col = colBase + waveN * 16 + l16;
  int o0  = (rowBase + waveM * 16 + half * 8) * ldo + col;
  float prev[8];
  if (acc) {
#pragma unroll
    for (int j = 0; j < 8; ++j) prev[j] = Out[o0 + j * ldo];
  }
  float bv = bias ? bias[col] : 0.0f;
#pragma unroll
  for (int j = 0; j < 8; ++j) {
    float v = cfrag[j] + bv;
    if (acc)  v += prev[j];
    if (relu) v = fmaxf(v, 0.0f);
    Out[o0 + j * ldo] = v;
  }
}

// ---------------- BatchNorm(+ReLU) over node axis: one block per channel ----
__global__ void k_bn_relu(float* x, int ldx, int M,
                          const float* __restrict__ g, const float* __restrict__ b) {
  int c = blockIdx.x;
  __shared__ float s1[256], s2[256];
  float sum = 0.0f, sq = 0.0f;
  for (int r = threadIdx.x; r < M; r += 256) {
    float v = x[(long)r * ldx + c];
    sum += v; sq += v * v;
  }
  s1[threadIdx.x] = sum; s2[threadIdx.x] = sq;
  __syncthreads();
  for (int s = 128; s > 0; s >>= 1) {
    if (threadIdx.x < s) { s1[threadIdx.x] += s1[threadIdx.x + s];
                           s2[threadIdx.x] += s2[threadIdx.x + s]; }
    __syncthreads();
  }
  float mean = s1[0] / M;
  float var  = s2[0] / M - mean * mean;
  float inv  = rsqrtf(var + 1e-5f);
  float gg = g[c], bb = b[c];
  for (int r = threadIdx.x; r < M; r += 256) {
    long i = (long)r * ldx + c;
    x[i] = fmaxf((x[i] - mean) * inv * gg + bb, 0.0f);
  }
}

// ---------------- AGNN pieces ----------------
__global__ void k_rownorm(const float* __restrict__ x, int ldx, int C,
                          float* __restrict__ xn, int M) {
  int r = blockIdx.x * blockDim.x + threadIdx.x;
  if (r >= M) return;
  const float* row = x + (long)r * ldx;
  float s = 0.0f;
  for (int c = 0; c < C; ++c) { float v = row[c]; s += v * v; }
  float inv = rsqrtf(fmaxf(s, 1e-24f));
  for (int c = 0; c < C; ++c) xn[(long)r * C + c] = row[c] * inv;
}

DEVINL unsigned f2mono(float f) {
  unsigned u = __float_as_uint(f);
  return (u & 0x80000000u) ? ~u : (u | 0x80000000u);
}
DEVINL float mono2f(unsigned u) {
  unsigned v = (u & 0x80000000u) ? (u & 0x7fffffffu) : ~u;
  return __uint_as_float(v);
}

__global__ void k_edge_logit(const float* __restrict__ xn, int C,
                             const int* __restrict__ src, const int* __restrict__ dst,
                             int n, int isLoop, const float* __restrict__ beta, int bidx,
                             float* __restrict__ logit, unsigned* __restrict__ mu) {
  int e = blockIdx.x * blockDim.x + threadIdx.x;
  if (e >= n) return;
  int s = isLoop ? e : src[e];
  int d = isLoop ? e : dst[e];
  const float4* a = (const float4*)(xn + (long)s * C);
  const float4* b = (const float4*)(xn + (long)d * C);
  float acc = 0.0f;
  for (int c = 0; c < (C >> 2); ++c) {
    float4 av = a[c], bv = b[c];
    acc += av.x * bv.x + av.y * bv.y + av.z * bv.z + av.w * bv.w;
  }
  float lg = beta[bidx] * acc;
  logit[e] = lg;
  atomicMax(mu + d, f2mono(lg));
}

__global__ void k_edge_exp(const float* __restrict__ logit,
                           const int* __restrict__ dst, int n, int isLoop,
                           const unsigned* __restrict__ mu,
                           float* __restrict__ wbuf, float* __restrict__ z) {
  int e = blockIdx.x * blockDim.x + threadIdx.x;
  if (e >= n) return;
  int d = isLoop ? e : dst[e];
  float w = __expf(logit[e] - mono2f(mu[d]));
  wbuf[e] = w;
  atomicAdd(z + d, w);
}

__global__ void k_edge_scatter(const float* __restrict__ x, int ldx, int C,
                               const int* __restrict__ src, const int* __restrict__ dst,
                               int n, int isLoop, const float* __restrict__ wbuf,
                               const float* __restrict__ z,
                               float* __restrict__ out, int ldo) {
  int e = blockIdx.x * blockDim.x + threadIdx.x;
  if (e >= n) return;
  int s = isLoop ? e : src[e];
  int d = isLoop ? e : dst[e];
  float a = wbuf[e] / z[d];
  const float* xs = x + (long)s * ldx;
  float* od = out + (long)d * ldo;
  for (int c = 0; c < C; ++c) atomicAdd(od + c, a * xs[c]);
}

// ---------------- mean aggregation over edges ----------------
__global__ void k_agg(const float* __restrict__ x, int ldx, int C,
                      const int* __restrict__ src, const int* __restrict__ dst, int n,
                      float* __restrict__ sum, float* __restrict__ cnt) {
  int e = blockIdx.x * blockDim.x + threadIdx.x;
  if (e >= n) return;
  int s = src[e], d = dst[e];
  atomicAdd(cnt + d, 1.0f);
  const float* xs = x + (long)s * ldx;
  for (int c = 0; c < C; ++c) atomicAdd(sum + (long)d * C + c, xs[c]);
}

__global__ void k_agg_div(float* sum, const float* __restrict__ cnt, int C) {
  int n = blockIdx.x * blockDim.x + threadIdx.x;
  if (n >= NNODE) return;
  float inv = 1.0f / fmaxf(cnt[n], 1.0f);
  for (int c = 0; c < C; ++c) sum[(long)n * C + c] *= inv;
}

// ===========================================================================
extern "C" void kernel_launch(void* const* d_in, const int* in_sizes, int n_in,
                              void* d_out, int out_size, void* d_ws, size_t ws_size,
                              hipStream_t stream) {
  // ---- inputs (setup_inputs insertion order; params flattened depth-first) --
  const int*   labels = (const int*)  d_in[0];
  const float* fx     = (const float*)d_in[1];
  const float* fy     = (const float*)d_in[2];
  const float* s0     = (const float*)d_in[3];
  const float* s1     = (const float*)d_in[4];
  const float* s2     = (const float*)d_in[5];
  const int*   pe     = (const int*)  d_in[6];
  const int*   ne     = (const int*)  d_in[7];
  const int E = in_sizes[6] / 3;
  const int* pe_src = pe;         const int* pe_dst = pe + E;
  const int* ne_src = ne;         const int* ne_dst = ne + E;

  int pi = 8;
  auto F = [&](void) { return (const float*)d_in[pi++]; };
  const float* pm_w = F(); const float* pm_b = F(); const float* pm_g = F(); const float* pm_be = F();
  const float* beta_pre = F();
  const float* beta_pos = F();
  const float* beta_neg = F();
  const float *mp_w[2], *mp_b[2], *mp_g[2], *mp_be[2];
  const float *mn_w[2], *mn_b[2], *mn_g[2], *mn_be[2];
  for (int i = 0; i < 2; ++i) { mp_w[i]=F(); mp_b[i]=F(); mp_g[i]=F(); mp_be[i]=F(); }
  for (int i = 0; i < 2; ++i) { mn_w[i]=F(); mn_b[i]=F(); mn_g[i]=F(); mn_be[i]=F(); }
  const float *g_plw[3], *g_plb[3], *g_prw[3], *g_prb[3];
  const float *g_nlw[3], *g_nlb[3], *g_nrw[3], *g_nrb[3];
  const float *g_bng[3], *g_bnb[3];
  for (int i = 0; i < 3; ++i) {
    g_plw[i]=F(); g_plb[i]=F(); g_prw[i]=F(); g_prb[i]=F();
    g_nlw[i]=F(); g_nlb[i]=F(); g_nrw[i]=F(); g_nrb[i]=F();
    g_bng[i]=F(); g_bnb[i]=F();
  }
  const float* lr_w = F(); const float* lr_b = F();
  (void)n_in; (void)ws_size; (void)out_size;

  // ---- workspace layout (floats) ----
  float* wsf = (float*)d_ws;
  long off = 0;
  auto carve = [&](long n) { float* p = wsf + off; off += n; return p; };
  float* cnt    = carve(NNODE);
  float* coords = carve((long)NNODE * 4);
  float* p0     = carve((long)NNODE * 32);
  float* p1     = carve((long)NNODE * 64);
  float* p2     = carve((long)NNODE * 128);
  float* xa     = carve((long)NNODE * 512);
  float* xb     = carve((long)NNODE * 512);
  float* xc     = carve((long)NNODE * 512);
  float* xn     = carve((long)NNODE * 128);
  float* aggp   = carve((long)NNODE * 128);
  float* aggn   = carve((long)NNODE * 128);
  float* acnt   = carve(NNODE);
  float* mu_f   = carve(NNODE);          // monotone-uint segment max
  float* z      = carve(NNODE);
  long  EL = 2L * E + NNODE;
  float* elog   = carve(EL);
  float* ew     = carve(EL);
  unsigned* mu  = (unsigned*)mu_f;

  auto cdiv = [](long n, long b) { return (int)((n + b - 1) / b); };
  auto fill = [&](float* p, float v, long n) {
    k_fill<<<cdiv(n, 256), 256, 0, stream>>>(p, v, n);
  };
  auto gemm = [&](const float* A, int lda, const float* Wt, int ldw, int rowoff,
                  const float* bias, float* Out, int ldo, int K, int N,
                  int acc, int relu) {
    int blocks = (NNODE / TM) * (N / TN);
    k_gemm_wmma<<<blocks, 256, 0, stream>>>(
        A, lda, Wt + (long)rowoff * ldw, ldw, bias, Out, ldo, NNODE, K, N, acc, relu);
  };
  auto bn = [&](float* x, int ldx, int C, const float* g, const float* b) {
    k_bn_relu<<<C, 256, 0, stream>>>(x, ldx, NNODE, g, b);
  };

  struct Seg { const int* src; const int* dst; int n; int loop; };
  auto agnn = [&](const float* x, int ldx, int C, float* out, int ldo,
                  const Seg* segs, int nseg, const float* beta, int bidx) {
    k_rownorm<<<cdiv(NNODE, 256), 256, 0, stream>>>(x, ldx, C, xn, NNODE);
    fill(mu_f, 0.0f, NNODE);             // mono(-NaN) == 0: identity for max
    fill(z, 0.0f, NNODE);
    long o = 0;
    for (int i = 0; i < nseg; ++i) {
      k_edge_logit<<<cdiv(segs[i].n, 256), 256, 0, stream>>>(
          xn, C, segs[i].src, segs[i].dst, segs[i].n, segs[i].loop,
          beta, bidx, elog + o, mu);
      o += segs[i].n;
    }
    o = 0;
    for (int i = 0; i < nseg; ++i) {
      k_edge_exp<<<cdiv(segs[i].n, 256), 256, 0, stream>>>(
          elog + o, segs[i].dst, segs[i].n, segs[i].loop, mu, ew + o, z);
      o += segs[i].n;
    }
    o = 0;
    for (int i = 0; i < nseg; ++i) {
      k_edge_scatter<<<cdiv(segs[i].n, 256), 256, 0, stream>>>(
          x, ldx, C, segs[i].src, segs[i].dst, segs[i].n, segs[i].loop,
          ew + o, z, out, ldo);
      o += segs[i].n;
    }
  };
  auto aggmean = [&](const float* x, int ldx, int C, const int* src, const int* dst,
                     float* out) {
    fill(out, 0.0f, (long)NNODE * C);
    fill(acnt, 0.0f, NNODE);
    k_agg<<<cdiv(E, 256), 256, 0, stream>>>(x, ldx, C, src, dst, E, out, acnt);
    k_agg_div<<<cdiv(NNODE, 256), 256, 0, stream>>>(out, acnt, C);
  };

  // ============================== stage 0 ==================================
  fill(cnt, 0.0f, NNODE);
  fill(coords, 0.0f, (long)NNODE * 4);
  fill(p0, 0.0f, (long)NNODE * 32);
  fill(p1, 0.0f, (long)NNODE * 64);
  fill(p2, 0.0f, (long)NNODE * 128);
  k_pool<<<cdiv((long)BATCH * WID * HEI, 256), 256, 0, stream>>>(
      labels, fx, fy, s0, s1, s2, coords, p0, p1, p2, cnt);
  k_pool_div<<<cdiv(NNODE, 256), 256, 0, stream>>>(coords, p0, p1, p2, cnt);

  // pre_merger: [p0(32) | coords(4)] @ W(36x32) + b  -> xa, then BN+ReLU
  gemm(p0, 32, pm_w, 32, 0,  nullptr, xa, 32, 32, 32, 0, 0);
  gemm(coords, 4, pm_w, 32, 32, pm_b, xa, 32, 4, 32, 1, 0);
  bn(xa, 32, 32, pm_g, pm_be);

  fill(xb, 0.0f, (long)NNODE * 32);
  Seg segs0[3] = {{pe_src, pe_dst, E, 0}, {ne_src, ne_dst, E, 0},
                  {nullptr, nullptr, NNODE, 1}};
  agnn(xa, 32, 32, xb, 32, segs0, 3, beta_pre, 0);

  // gcn0: out_p = lin(agg(x,pe),pos_l)+lin(x,pos_r); out_n analog; x -> xc(128)
  aggmean(xb, 32, 32, pe_src, pe_dst, aggp);
  aggmean(xb, 32, 32, ne_src, ne_dst, aggn);
  gemm(aggp, 32, g_plw[0], 64, 0, g_plb[0], xc,      128, 32, 64, 0, 0);
  gemm(xb,   32, g_prw[0], 64, 0, g_prb[0], xc,      128, 32, 64, 1, 0);
  gemm(aggn, 32, g_nlw[0], 64, 0, g_nlb[0], xc + 64, 128, 32, 64, 0, 0);
  gemm(xb,   32, g_nrw[0], 64, 0, g_nrb[0], xc + 64, 128, 32, 64, 1, 0);
  bn(xc, 128, 128, g_bng[0], g_bnb[0]);

  Seg segp[2] = {{pe_src, pe_dst, E, 0}, {nullptr, nullptr, NNODE, 1}};
  Seg segn[2] = {{ne_src, ne_dst, E, 0}, {nullptr, nullptr, NNODE, 1}};

  // ============================== stage 1 ==================================
  // x_pos = xc[:,64:], x_neg = xc[:,:64]
  gemm(xc + 64, 128, mp_w[0], 64, 0,   nullptr,  xa, 64, 64, 64, 0, 0);
  gemm(p1,      64,  mp_w[0], 64, 64,  nullptr,  xa, 64, 64, 64, 1, 0);
  gemm(coords,  4,   mp_w[0], 64, 128, mp_b[0],  xa, 64, 4,  64, 1, 0);
  bn(xa, 64, 64, mp_g[0], mp_be[0]);
  fill(xb, 0.0f, (long)NNODE * 128);
  agnn(xa, 64, 64, xb, 128, segp, 2, beta_pos, 0);

  gemm(xc,     128, mn_w[0], 64, 0,   nullptr,  xa, 64, 64, 64, 0, 0);
  gemm(p1,     64,  mn_w[0], 64, 64,  nullptr,  xa, 64, 64, 64, 1, 0);
  gemm(coords, 4,   mn_w[0], 64, 128, mn_b[0],  xa, 64, 4,  64, 1, 0);
  bn(xa, 64, 64, mn_g[0], mn_be[0]);
  agnn(xa, 64, 64, xb + 64, 128, segn, 2, beta_neg, 0);

  // gcn1: xp = xb[:,:64], xg = xb[:,64:]
  aggmean(xb,      128, 64, pe_src, pe_dst, aggp);     // agg(xp, pe)
  aggmean(xb + 64, 128, 64, ne_src, ne_dst, aggn);     // agg(xg, ne)
  gemm(aggp, 64, g_plw[1], 128, 0,  g_plb[1], xc,       256, 64, 128, 0, 0);
  gemm(aggn, 64, g_plw[1], 128, 64, nullptr,  xc,       256, 64, 128, 1, 0);
  gemm(xb,  128, g_prw[1], 128, 0,  g_prb[1], xc,       256, 64, 128, 1, 0);
  aggmean(xb + 64, 128, 64, pe_src, pe_dst, aggp);     // agg(xg, pe)
  aggmean(xb,      128, 64, ne_src, ne_dst, aggn);     // agg(xp, ne)
  gemm(aggp, 64, g_nlw[1], 128, 0,  g_nlb[1], xc + 128, 256, 64, 128, 0, 0);
  gemm(aggn, 64, g_nlw[1], 128, 64, nullptr,  xc + 128, 256, 64, 128, 1, 0);
  gemm(xb + 64, 128, g_nrw[1], 128, 0, g_nrb[1], xc + 128, 256, 64, 128, 1, 0);
  bn(xc, 256, 256, g_bng[1], g_bnb[1]);

  // ============================== stage 2 ==================================
  gemm(xc + 128, 256, mp_w[1], 128, 0,   nullptr, xa, 128, 128, 128, 0, 0);
  gemm(p2,       128, mp_w[1], 128, 128, nullptr, xa, 128, 128, 128, 1, 0);
  gemm(coords,   4,   mp_w[1], 128, 256, mp_b[1], xa, 128, 4,   128, 1, 0);
  bn(xa, 128, 128, mp_g[1], mp_be[1]);
  fill(xb, 0.0f, (long)NNODE * 256);
  agnn(xa, 128, 128, xb, 256, segp, 2, beta_pos, 1);

  gemm(xc,     256, mn_w[1], 128, 0,   nullptr, xa, 128, 128, 128, 0, 0);
  gemm(p2,     128, mn_w[1], 128, 128, nullptr, xa, 128, 128, 128, 1, 0);
  gemm(coords, 4,   mn_w[1], 128, 256, mn_b[1], xa, 128, 4,   128, 1, 0);
  bn(xa, 128, 128, mn_g[1], mn_be[1]);
  agnn(xa, 128, 128, xb + 128, 256, segn, 2, beta_neg, 1);

  // gcn2: xp = xb[:,:128], xg = xb[:,128:]
  aggmean(xb,       256, 128, pe_src, pe_dst, aggp);
  aggmean(xb + 128, 256, 128, ne_src, ne_dst, aggn);
  gemm(aggp, 128, g_plw[2], 256, 0,   g_plb[2], xc,       512, 128, 256, 0, 0);
  gemm(aggn, 128, g_plw[2], 256, 128, nullptr,  xc,       512, 128, 256, 1, 0);
  gemm(xb,   256, g_prw[2], 256, 0,   g_prb[2], xc,       512, 128, 256, 1, 0);
  aggmean(xb + 128, 256, 128, pe_src, pe_dst, aggp);
  aggmean(xb,       256, 128, ne_src, ne_dst, aggn);
  gemm(aggp, 128, g_nlw[2], 256, 0,   g_nlb[2], xc + 256, 512, 128, 256, 0, 0);
  gemm(aggn, 128, g_nlw[2], 256, 128, nullptr,  xc + 256, 512, 128, 256, 1, 0);
  gemm(xb + 128, 256, g_nrw[2], 256, 0, g_nrb[2], xc + 256, 512, 128, 256, 1, 0);
  bn(xc, 512, 512, g_bng[2], g_bnb[2]);

  // ====================== lin_reduc + outputs ==============================
  gemm(xc, 512, lr_w, 256, 0, lr_b, (float*)d_out, 256, 512, 256, 0, 1);

  int* out_i = (int*)d_out + (long)NNODE * 256;
  k_copy_i32<<<cdiv(3L * E, 256), 256, 0, stream>>>(pe, out_i, 3L * E);
  k_copy_i32<<<cdiv(3L * E, 256), 256, 0, stream>>>(ne, out_i + 3L * E, 3L * E);
}
